// RNN_55929064128966
// MI455X (gfx1250) — matrix operand order repaired and checked by reading, not compiled, not run on hardware
//
#include <hip/hip_runtime.h>

typedef _Float16 v16h __attribute__((ext_vector_type(16)));
typedef _Float16 v8h  __attribute__((ext_vector_type(8)));
typedef float    v8f  __attribute__((ext_vector_type(8)));
typedef float    v4f  __attribute__((ext_vector_type(4)));

#define B_ 64
#define T_ 2048
#define D_ 256
#define H_ 256

__device__ inline v8f splat8(float v) {
  v8f r;
#pragma unroll
  for (int i = 0; i < 8; ++i) r[i] = v;
  return r;
}

__device__ inline v16h cat16(v8h lo, v8h hi) {
  return __builtin_shufflevector(lo, hi, 0, 1, 2, 3, 4, 5, 6, 7, 8, 9, 10, 11,
                                 12, 13, 14, 15);
}

// gfx1250 async global->LDS copy of 16 bytes (ASYNCcnt-tracked).
// lds_off: workgroup-relative LDS byte offset (low 32 bits of generic ptr).
__device__ inline void async_copy_b128(unsigned lds_off, const void* gaddr) {
  asm volatile("global_load_async_to_lds_b128 %0, %1, off"
               :
               : "v"(lds_off), "v"(gaddr)
               : "memory");
}

__device__ inline void wait_async0() {
  asm volatile("s_wait_asynccnt 0x0" ::: "memory");
}

// ---------------------------------------------------------------------------
// Kernel 0: transpose-convert W (DxH) and W_recurrent (HxH) into f16 [N][K]
// so that WMMA B-fragments become two 16-byte per-lane loads.
// ---------------------------------------------------------------------------
__global__ void convert_weights(const float* __restrict__ W,
                                const float* __restrict__ Wr,
                                _Float16* __restrict__ WhT,
                                _Float16* __restrict__ WrT) {
  const int n = blockIdx.x;   // column of source / row of dest
  const int k = threadIdx.x;  // row of source
  WhT[n * D_ + k] = (_Float16)W[k * H_ + n];
  WrT[n * H_ + k] = (_Float16)Wr[k * H_ + n];
}

// ---------------------------------------------------------------------------
// Kernel 1: xW = x @ W + b  as f16 into workspace.
// Block = 8 waves; all waves share one 64-column slab of WhT staged into LDS
// via async global->LDS (32 KB). Each wave computes a 16(M) x 64(N) tile:
// all 8 A fragments preloaded from global, B fragments double-buffered from
// LDS across the K loop.
// ---------------------------------------------------------------------------
__global__ void __launch_bounds__(256)
xw_gemm(const float* __restrict__ x, const _Float16* __restrict__ WhT,
        const float* __restrict__ bias, _Float16* __restrict__ xwh) {
  __shared__ _Float16 Bs[64 * 256];  // 32 KB: rows = local N 0..63, cols = K

  const int nq   = blockIdx.x & 3;   // 64-column slab
  const int rblk = blockIdx.x >> 2;  // 8 row tiles per block
  const int lane = threadIdx.x & 31;
  const int wav  = threadIdx.x >> 5;  // 0..7
  const int mrow = lane & 15;
  const int hi   = lane >> 4;
  const int kbA  = hi * 8;   // 16-bit A layout K base
  const int kbB  = hi * 8;   // 16-bit B layout K base
  const int mC   = hi * 8;   // C/D row base

  // ---- async-stage the B slab (64 rows x 256 K, contiguous 32 KB) ----
  {
    const _Float16* src = WhT + (size_t)(nq * 64) * D_;
    const unsigned lds0 = (unsigned)(uintptr_t)(&Bs[0]);
#pragma unroll
    for (int i = 0; i < 8; ++i) {
      const int e = (i * 256 + threadIdx.x) * 8;  // halves, 16B chunks
      async_copy_b128(lds0 + e * 2, src + e);
    }
    wait_async0();
  }

  // ---- preload all 8 A fragments (f32 -> f16) while staging completes ----
  const int rtile = rblk * 8 + wav;
  const float* xrow = x + (size_t)(rtile * 16 + mrow) * D_;
  v16h a[8];
#pragma unroll
  for (int kc = 0; kc < 8; ++kc) {
    const float* ap = xrow + kc * 32 + kbA;
    v4f f0 = *(const v4f*)(ap + 0);
    v4f f1 = *(const v4f*)(ap + 4);
    v4f f2 = *(const v4f*)(ap + 16);
    v4f f3 = *(const v4f*)(ap + 20);
#pragma unroll
    for (int i = 0; i < 4; ++i) {
      a[kc][i]      = (_Float16)f0[i];
      a[kc][4 + i]  = (_Float16)f1[i];
      a[kc][8 + i]  = (_Float16)f2[i];
      a[kc][12 + i] = (_Float16)f3[i];
    }
  }

  v8f c[4];
#pragma unroll
  for (int j = 0; j < 4; ++j)
    c[j] = splat8(bias[nq * 64 + j * 16 + mrow]);

  __syncthreads();  // B slab visible to all waves

  // ---- K loop: B fragments from LDS, double-buffered in registers ----
  const _Float16* bs = &Bs[0];
  v16h bf[4], nbf[4];
#pragma unroll
  for (int j = 0; j < 4; ++j) {
    const _Float16* bp = bs + (j * 16 + mrow) * 256 + kbB;
    bf[j] = cat16(*(const v8h*)bp, *(const v8h*)(bp + 16));
  }
#pragma unroll
  for (int kc = 0; kc < 8; ++kc) {
    if (kc < 7) {
#pragma unroll
      for (int j = 0; j < 4; ++j) {
        const _Float16* bp = bs + (j * 16 + mrow) * 256 + (kc + 1) * 32 + kbB;
        nbf[j] = cat16(*(const v8h*)bp, *(const v8h*)(bp + 16));
      }
    }
#pragma unroll
    for (int j = 0; j < 4; ++j)
      c[j] = __builtin_amdgcn_wmma_f32_16x16x32_f16(false, a[kc], false, bf[j],
                                                    (short)0, c[j], false,
                                                    false);
#pragma unroll
    for (int j = 0; j < 4; ++j) bf[j] = nbf[j];
  }

#pragma unroll
  for (int j = 0; j < 4; ++j) {
    const int col = nq * 64 + j * 16 + mrow;
#pragma unroll
    for (int r = 0; r < 8; ++r) {
      const int row = rtile * 16 + mC + r;
      xwh[(size_t)row * H_ + col] = (_Float16)c[j][r];
    }
  }
}

// ---------------------------------------------------------------------------
// Kernel 2: sequential scan  h = tanh(xw_t + h @ Wr).
// One block per 16-batch tile (4 blocks), 16 waves per block, each wave owns
// a 16-column N-tile. Wr B-fragments live in VGPRs; h is double-buffered f16
// in LDS with one barrier per timestep. xw tile loads are software-pipelined
// one timestep ahead so they hide behind the WMMA chain + barrier.
// ---------------------------------------------------------------------------
__global__ void __launch_bounds__(512)
rnn_scan(const _Float16* __restrict__ xwh, const _Float16* __restrict__ WrT,
         float* __restrict__ out) {
  __shared__ _Float16 hbuf[2][16 * H_];  // 16 KB

  const int lane = threadIdx.x & 31;
  const int wav  = threadIdx.x >> 5;  // 0..15
  const int n0   = wav * 16;
  const int b0   = blockIdx.x * 16;
  const int mrow = lane & 15;
  const int hi   = lane >> 4;
  const int kbA  = hi * 8;
  const int kbB  = hi * 8;
  const int mC   = hi * 8;

  // Pin all 8 K-chunk B-fragments of Wr for this wave's N-tile in registers.
  v16h wr[8];
  {
    const _Float16* base = WrT + (n0 + mrow) * H_ + kbB;
#pragma unroll
    for (int kc = 0; kc < 8; ++kc) {
      const _Float16* bp = base + kc * 32;
      wr[kc] = cat16(*(const v8h*)bp, *(const v8h*)(bp + 16));
    }
  }

  // h0 = 0
  for (int i = threadIdx.x; i < 16 * H_; i += 512) {
    hbuf[0][i] = (_Float16)0.0f;
    hbuf[1][i] = (_Float16)0.0f;
  }

  // Software-pipelined xw tile: preload t=0.
  _Float16 xv[8];
#pragma unroll
  for (int r = 0; r < 8; ++r) {
    const int row = b0 + mC + r;  // batch index
    xv[r] = xwh[((size_t)row * T_ + 0) * H_ + n0 + mrow];
  }

  __syncthreads();

  int p = 0;
  for (int t = 0; t < T_; ++t) {
    // C init from the pipelined registers, then issue loads for t+1.
    v8f c;
#pragma unroll
    for (int r = 0; r < 8; ++r) c[r] = (float)xv[r];
    if (t + 1 < T_) {
#pragma unroll
      for (int r = 0; r < 8; ++r) {
        const int row = b0 + mC + r;
        xv[r] = xwh[((size_t)row * T_ + (t + 1)) * H_ + n0 + mrow];
      }
    }

    // h_tile @ Wr : 8 WMMAs over K=256, A fragments from LDS.
    const _Float16* hb = &hbuf[p][0];
#pragma unroll
    for (int kc = 0; kc < 8; ++kc) {
      const _Float16* ap = hb + mrow * H_ + kc * 32 + kbA;
      v16h af = cat16(*(const v8h*)ap, *(const v8h*)(ap + 16));
      c = __builtin_amdgcn_wmma_f32_16x16x32_f16(false, af, false, wr[kc],
                                                 (short)0, c, false, false);
    }

    // tanh, stream result out (NT), and write new h tile into other buffer.
    _Float16* hn = &hbuf[1 - p][0];
#pragma unroll
    for (int r = 0; r < 8; ++r) {
      float cv = c[r];
      cv = fminf(fmaxf(cv, -15.0f), 15.0f);
      const float e  = __builtin_amdgcn_exp2f(cv * 2.8853900817779268f);
      const float th = (e - 1.0f) * __builtin_amdgcn_rcpf(e + 1.0f);
      const int row = b0 + mC + r;
      __builtin_nontemporal_store(th,
                                  &out[((size_t)row * T_ + t) * H_ + n0 + mrow]);
      hn[(mC + r) * H_ + n0 + mrow] = (_Float16)th;
    }
    __syncthreads();
    p ^= 1;
  }
}

// ---------------------------------------------------------------------------
extern "C" void kernel_launch(void* const* d_in, const int* in_sizes, int n_in,
                              void* d_out, int out_size, void* d_ws,
                              size_t ws_size, hipStream_t stream) {
  const float* x  = (const float*)d_in[0];
  const float* W  = (const float*)d_in[1];
  const float* Wr = (const float*)d_in[2];
  const float* b  = (const float*)d_in[3];
  float* out = (float*)d_out;

  char* ws = (char*)d_ws;
  _Float16* WhT = (_Float16*)ws;                              // 128 KB
  _Float16* WrT = (_Float16*)(ws + (size_t)D_ * H_ * 2);      // 128 KB
  _Float16* xwh = (_Float16*)(ws + (size_t)2 * D_ * H_ * 2);  // 64 MB

  convert_weights<<<256, 256, 0, stream>>>(W, Wr, WhT, WrT);

  // 1024 row-blocks (8 tiles each) x 4 column slabs
  xw_gemm<<<4096, 256, 0, stream>>>(x, WhT, b, xwh);

  rnn_scan<<<4, 512, 0, stream>>>(xwh, WrT, out);
}
